// GAT_21930103013469
// MI455X (gfx1250) — matrix operand order repaired and checked by reading, not compiled, not run on hardware
//
#include <hip/hip_runtime.h>
#include <hip/hip_bf16.h>
#include <stdint.h>

// ---------------------------------------------------------------------------
// GAT layer for MI455X (gfx1250, wave32, WMMA).
// B=16, N=2048, Fin=Fout=256.
// ---------------------------------------------------------------------------

typedef __attribute__((ext_vector_type(16))) __bf16 v16bf;
typedef __attribute__((ext_vector_type(8)))  __bf16 v8bf;
typedef __attribute__((ext_vector_type(8)))  float  v8f;
typedef __attribute__((ext_vector_type(4)))  float  fltv4;
typedef __attribute__((ext_vector_type(4)))  int    intv4;

#define B_    16
#define N_    2048
#define F_    256
#define ALPHA 0.2f

static __device__ __forceinline__ float lrelu(float x) { return fmaxf(x, ALPHA * x); }

// ---------------------------------------------------------------------------
// Kernel 1: WhbT[b][o][i] = bf16( sum_k h[b][i][k] * W[b][k][o] )
// One wave per 16x16 output tile, K-loop of 8 x (16x16x32 bf16 WMMA).
// C/D layout (lane holds col o, rows r+8*half) -> transposed store is a
// single contiguous 16B store per lane.
// ---------------------------------------------------------------------------
__global__ __launch_bounds__(256) void k1_gemm_wh(const float* __restrict__ h,
                                                  const float* __restrict__ W,
                                                  __bf16* __restrict__ whbT) {
    const int lane = threadIdx.x & 31;
    const int wv   = threadIdx.x >> 5;
    const int t    = blockIdx.x * 8 + wv;          // 32768 tiles total
    const int b    = t >> 11;                      // 2048 tiles per batch
    const int r    = t & 2047;
    const int i0   = (r >> 4) << 4;                // row tile
    const int o0   = (r & 15) << 4;                // col tile
    const int half = lane >> 4;
    const int l15  = lane & 15;

    const float* hrow = h + ((size_t)b * N_ + (i0 + l15)) * F_;
    const float* wb   = W + (size_t)b * F_ * F_;
    const int    oB   = o0 + l15;

    v8f acc = {0.f, 0.f, 0.f, 0.f, 0.f, 0.f, 0.f, 0.f};

    for (int k0 = 0; k0 < F_; k0 += 32) {
        // A fragment: lane holds row i, K runs {half*8..+7} and {16+half*8..+7}
        const float* pa = hrow + k0 + half * 8;
        float bufA[16];
        *(fltv4*)&bufA[0]  = *(const fltv4*)(pa + 0);
        *(fltv4*)&bufA[4]  = *(const fltv4*)(pa + 4);
        *(fltv4*)&bufA[8]  = *(const fltv4*)(pa + 16);
        *(fltv4*)&bufA[12] = *(const fltv4*)(pa + 20);
        v16bf af;
#pragma unroll
        for (int e = 0; e < 16; ++e) af[e] = (__bf16)bufA[e];

        // B fragment: lane holds col o, K runs {half*8..+7} and {16+half*8..+7} (strided in mem)
        const int kb = k0 + half * 8;
        v16bf bf;
#pragma unroll
        for (int e = 0; e < 16; ++e) bf[e] = (__bf16)wb[(size_t)(kb + (e & 7) + ((e >> 3) << 4)) * F_ + oB];

        acc = __builtin_amdgcn_wmma_f32_16x16x32_bf16(false, af, false, bf,
                                                      (short)0, acc, false, false);
    }

    // Transposed store: WhbT[b][oB][i0 + half*8 + r], 8 bf16 = 16B contiguous.
    const int ibase = i0 + half * 8;
    v8bf sv;
#pragma unroll
    for (int q = 0; q < 8; ++q) sv[q] = (__bf16)acc[q];
    *(v8bf*)(whbT + ((size_t)b * F_ + oB) * N_ + ibase) = sv;
}

// ---------------------------------------------------------------------------
// Kernel 2: Wh1[b][i] = sum_o WhbT[b][o][i]*a[b][o]; Wh2 with a[b][256+o].
// Threads map to i (coalesced along i for every o).
// ---------------------------------------------------------------------------
__global__ __launch_bounds__(256) void k2_rowvec(const __bf16* __restrict__ whbT,
                                                 const float* __restrict__ a,
                                                 float* __restrict__ wh1,
                                                 float* __restrict__ wh2) {
    __shared__ float sa[2 * F_];
    const int tid = threadIdx.x;
    const int b   = blockIdx.y;
    for (int k = tid; k < 2 * F_; k += 256) sa[k] = a[(size_t)b * 2 * F_ + k];
    __syncthreads();

    const int i = blockIdx.x * 256 + tid;
    float s1 = 0.f, s2 = 0.f;
#pragma unroll 4
    for (int o = 0; o < F_; ++o) {
        float v = (float)whbT[((size_t)b * F_ + o) * N_ + i];
        s1 += v * sa[o];
        s2 += v * sa[F_ + o];
    }
    wh1[(size_t)b * N_ + i] = s1;
    wh2[(size_t)b * N_ + i] = s2;
}

// ---------------------------------------------------------------------------
// Kernel 3: rmax[b] = max_i Wh1[b][i]  (safe-softmax upper bound)
// ---------------------------------------------------------------------------
__global__ __launch_bounds__(256) void k3_rowmax(const float* __restrict__ wh1,
                                                 float* __restrict__ rmax) {
    __shared__ float sm[256];
    const int b = blockIdx.x, tid = threadIdx.x;
    float m = -1e30f;
    for (int i = tid; i < N_; i += 256) m = fmaxf(m, wh1[(size_t)b * N_ + i]);
    sm[tid] = m;
    __syncthreads();
    for (int s = 128; s > 0; s >>= 1) {
        if (tid < s) sm[tid] = fmaxf(sm[tid], sm[tid + s]);
        __syncthreads();
    }
    if (tid == 0) rmax[b] = sm[0];
}

// ---------------------------------------------------------------------------
// Kernel 4: column softmax stats (softmax over axis=1 -> over i).
//   m[b][j]  = lrelu(rmax[b] + Wh2[b][j])   (upper bound of masked max)
//   rz[b][j] = 1 / sum_i adj ? exp(lrelu(Wh1[i]+Wh2[j]) - m[j]) : 0
// adj read coalesced along j (thread dim).
// ---------------------------------------------------------------------------
__global__ __launch_bounds__(256) void k4_colsoft(const int* __restrict__ adj,
                                                  const float* __restrict__ wh1,
                                                  const float* __restrict__ wh2,
                                                  const float* __restrict__ rmax,
                                                  float* __restrict__ cm,
                                                  float* __restrict__ cr) {
    __shared__ float sw[N_];
    const int tid = threadIdx.x;
    const int b   = blockIdx.y;
    for (int i = tid; i < N_; i += 256) sw[i] = wh1[(size_t)b * N_ + i];
    __syncthreads();

    const int   j  = blockIdx.x * 256 + tid;
    const float w2 = wh2[(size_t)b * N_ + j];
    const float mj = lrelu(rmax[b] + w2);
    const int*  ac = adj + (size_t)b * N_ * N_ + j;

    float Z = 0.f;
#pragma unroll 4
    for (int i = 0; i < N_; ++i) {
        int   av = ac[(size_t)i * N_];
        float e  = lrelu(sw[i] + w2) - mj;
        Z += (av > 0) ? __expf(e) : 0.f;
    }
    cm[(size_t)b * N_ + j] = mj;
    cr[(size_t)b * N_ + j] = 1.0f / fmaxf(Z, 1e-30f);
}

// ---------------------------------------------------------------------------
// Kernel 5: fused attention matmul + ELU.
//   out[b][i][:] = ELU( sum_j P[i][j] * Wh[b][j][:] ),
//   P[i][j] = adj ? exp(lrelu(Wh1[i]+Wh2[j]) - m[j]) * rz[j] : 0   (bf16)
// Block = 64 rows x 256 cols, 8 waves (4 row-groups x 2 col-groups).
// Per K-step (32 j's): async-stage WhbT 32x256 tile to LDS (double-buffered
// via global_load_async_to_lds_b128 + s_wait_asynccnt), build P A-fragment in
// registers in the documented lane layout, 8 WMMAs per wave.
// ---------------------------------------------------------------------------
#define LROW 40  // LDS row stride in bf16 (32 data + pad), 80B: 16B aligned, conflict-light

__global__ __launch_bounds__(256) void k5_attn(const int* __restrict__ adj,
                                               const __bf16* __restrict__ whbT,
                                               const float* __restrict__ wh1,
                                               const float* __restrict__ wh2,
                                               const float* __restrict__ cm,
                                               const float* __restrict__ cr,
                                               float* __restrict__ out) {
    __shared__ __align__(16) __bf16 Bs[2][256 * LROW];

    const int tid  = threadIdx.x;
    const int lane = tid & 31;
    const int wv   = tid >> 5;
    const int b    = blockIdx.x >> 5;   // 32 row-tiles per batch
    const int tile = blockIdx.x & 31;
    const int i0   = tile * 64;
    const int rg   = wv & 3;            // row group (16 rows each)
    const int cg   = wv >> 2;           // col group (128 cols each)
    const int half = lane >> 4;
    const int l15  = lane & 15;

    const int    iRow = i0 + rg * 16 + l15;
    const float  wh1i = wh1[(size_t)b * N_ + iRow];
    const int*   arow = adj + ((size_t)b * N_ + iRow) * N_;
    const float* pw2  = wh2 + (size_t)b * N_;
    const float* pm   = cm + (size_t)b * N_;
    const float* pr   = cr + (size_t)b * N_;
    const __bf16* wsrc = whbT + (size_t)b * F_ * N_;   // [o][j]

    v8f acc[8];
#pragma unroll
    for (int t = 0; t < 8; ++t) acc[t] = (v8f){0.f, 0.f, 0.f, 0.f, 0.f, 0.f, 0.f, 0.f};

    // --- async stage of one 32x256 WhbT K-tile into LDS (row o -> 64B) ------
    auto stage = [&](int js) {
        const int      buf = js & 1;
        const int      o   = tid;                       // 256 threads = 256 rows
        const __bf16*  gp  = wsrc + (size_t)o * N_ + js * 32;
        const uint32_t lds = (uint32_t)(uintptr_t)(&Bs[buf][o * LROW]);
#pragma unroll
        for (int s = 0; s < 4; ++s) {
            asm volatile("global_load_async_to_lds_b128 %0, %1, off"
                         :: "v"(lds + s * 16), "v"(gp + s * 8)
                         : "memory");
        }
    };

    const int NS = N_ / 32;
    stage(0);

    for (int js = 0; js < NS; ++js) {
        if (js + 1 < NS) {
            stage(js + 1);                               // prefetch next tile
            asm volatile("s_wait_asynccnt 0x4" ::: "memory");  // tile js done
        } else {
            asm volatile("s_wait_asynccnt 0x0" ::: "memory");
        }
        __syncthreads();

        // ---- build P A-fragment (16x32) in registers ----------------------
        const int jA = js * 32 + half * 8;               // run A; run B = +16
        intv4 qa[4];
        fltv4 qw[4], qm[4], qr[4];
        qa[0] = *(const intv4*)(arow + jA);      qa[1] = *(const intv4*)(arow + jA + 4);
        qa[2] = *(const intv4*)(arow + jA + 16); qa[3] = *(const intv4*)(arow + jA + 20);
        qw[0] = *(const fltv4*)(pw2 + jA);       qw[1] = *(const fltv4*)(pw2 + jA + 4);
        qw[2] = *(const fltv4*)(pw2 + jA + 16);  qw[3] = *(const fltv4*)(pw2 + jA + 20);
        qm[0] = *(const fltv4*)(pm + jA);        qm[1] = *(const fltv4*)(pm + jA + 4);
        qm[2] = *(const fltv4*)(pm + jA + 16);   qm[3] = *(const fltv4*)(pm + jA + 20);
        qr[0] = *(const fltv4*)(pr + jA);        qr[1] = *(const fltv4*)(pr + jA + 4);
        qr[2] = *(const fltv4*)(pr + jA + 16);   qr[3] = *(const fltv4*)(pr + jA + 20);

        v16bf A;
#pragma unroll
        for (int g = 0; g < 4; ++g) {
#pragma unroll
            for (int e = 0; e < 4; ++e) {
                float x = lrelu(wh1i + qw[g][e]) - qm[g][e];
                float p = (qa[g][e] > 0) ? __expf(x) * qr[g][e] : 0.f;
                A[g * 4 + e] = (__bf16)p;
            }
        }

        // ---- 8 WMMAs: B frag = 2 contiguous 16B LDS reads per lane --------
        const __bf16* lbase = &Bs[js & 1][0];
#pragma unroll
        for (int t = 0; t < 8; ++t) {
            const int     o  = cg * 128 + t * 16 + l15;
            const __bf16* lp = lbase + o * LROW + half * 8;
            v16bf Bf;
            __builtin_memcpy(&Bf, lp, 16);
            __builtin_memcpy((char*)&Bf + 16, lp + 16, 16);
            acc[t] = __builtin_amdgcn_wmma_f32_16x16x32_bf16(false, A, false, Bf,
                                                             (short)0, acc[t], false, false);
        }
        __syncthreads();   // before next stage overwrites the other buffer's peer
    }

    // ---- ELU + store ------------------------------------------------------
    const int ibase = i0 + rg * 16 + half * 8;
#pragma unroll
    for (int t = 0; t < 8; ++t) {
        const int o = cg * 128 + t * 16 + l15;
        float* orow = out + ((size_t)b * N_ + ibase) * F_ + o;
#pragma unroll
        for (int q = 0; q < 8; ++q) {
            float v = acc[t][q];
            v = (v > 0.f) ? v : (__expf(v) - 1.0f);
            orow[(size_t)q * F_] = v;
        }
    }
}

// ---------------------------------------------------------------------------
// Launch
// ---------------------------------------------------------------------------
extern "C" void kernel_launch(void* const* d_in, const int* in_sizes, int n_in,
                              void* d_out, int out_size, void* d_ws, size_t ws_size,
                              hipStream_t stream) {
    const float* h   = (const float*)d_in[0];
    const int*   adj = (const int*)d_in[1];
    const float* W   = (const float*)d_in[2];
    const float* a   = (const float*)d_in[3];
    float*       out = (float*)d_out;

    char* ws = (char*)d_ws;
    __bf16* whbT = (__bf16*)(ws);                                   // 16 MB
    float*  wh1  = (float*)(ws + 16777216);                         // 128 KB
    float*  wh2  = (float*)(ws + 16777216 + 131072);                // 128 KB
    float*  cm   = (float*)(ws + 16777216 + 2 * 131072);            // 128 KB
    float*  cr   = (float*)(ws + 16777216 + 3 * 131072);            // 128 KB
    float*  rmax = (float*)(ws + 16777216 + 4 * 131072);            // 64 B

    k1_gemm_wh<<<4096, 256, 0, stream>>>(h, W, whbT);
    k2_rowvec<<<dim3(N_ / 256, B_), 256, 0, stream>>>(whbT, a, wh1, wh2);
    k3_rowmax<<<B_, 256, 0, stream>>>(wh1, rmax);
    k4_colsoft<<<dim3(N_ / 256, B_), 256, 0, stream>>>(adj, wh1, wh2, rmax, cm, cr);
    k5_attn<<<B_ * (N_ / 64), 256, 0, stream>>>(adj, whbT, wh1, wh2, cm, cr, out);
}
